// Block_k_51642686767351
// MI455X (gfx1250) — matrix operand order repaired and checked by reading, not compiled, not run on hardware
//
#include <hip/hip_runtime.h>
#include <hip/hip_bf16.h>

// ---------------------------------------------------------------------------
// Transformer block (B=8,N=1024,C=768,H=12,HS=64) for MI455X / gfx1250.
// GEMM-shaped math on v_wmma_f32_16x16x32_f16 (f16 in, f32 acc).
// Tile staging via the Tensor Data Mover (tensor_load_to_lds + s_wait_tensorcnt)
// with LDS padding done by the TDM descriptor; cooperative-load fallback kept.
// Workspace usage ~120 MB.
// ---------------------------------------------------------------------------

typedef __attribute__((ext_vector_type(16))) _Float16 v16h;
typedef __attribute__((ext_vector_type(8)))  float    v8f;
typedef __attribute__((ext_vector_type(4)))  unsigned v4u;
typedef __attribute__((ext_vector_type(8)))  int      v8i;
typedef __attribute__((ext_vector_type(4)))  int      v4i;

union F16x16 {
    v16h     v;
    _Float16 h[16];
    uint4    q[2];
};

#define WMMA_F16(A, Bf, Cc) \
    __builtin_amdgcn_wmma_f32_16x16x32_f16(false, (A), false, (Bf), (short)0, (Cc), false, false)

// ------------------------- Tensor Data Mover helpers ------------------------
#if defined(__gfx1250__) && __has_builtin(__builtin_amdgcn_tensor_load_to_lds)
#define USE_TDM 1

__device__ __forceinline__ unsigned lds_offset_of(const void* p) {
    // __shared__ objects live in AS(3); ptrtoint of the AS3 pointer is the
    // byte offset TDM wants in D#.lds_addr.
    return (unsigned)(unsigned long long)(__attribute__((address_space(3))) const char*)p;
}

// 2D tile load: tile_w elements (data_size=2B) x tile_h rows, row stride
// `stride_elems` in global memory, optional LDS padding (ISA 8.4 group1 codes:
// pad after 2^(pi+1) DWORDs, amount pa+1 DWORDs).
__device__ __forceinline__ void tdm_load_2d(unsigned lds_byte, const void* gptr,
                                            unsigned tile_w, unsigned tile_h,
                                            unsigned stride_elems,
                                            unsigned pad_interval_code,
                                            unsigned pad_amount_code) {
    const unsigned long long ga = (unsigned long long)gptr;
    v4u g0;
    g0[0] = 1u;                                          // count=1, user mode
    g0[1] = lds_byte;                                    // D#.lds_addr
    g0[2] = (unsigned)(ga & 0xffffffffu);                // global_addr[31:0]
    g0[3] = (unsigned)((ga >> 32) & 0x01ffffffu) | (2u << 30);  // [56:32] | type=2
    v8i g1;
    g1[0] = (int)((1u << 16)                             // data_size = 2 bytes
                  | (1u << 20)                           // pad_enable
                  | (pad_interval_code << 22)
                  | (pad_amount_code << 25));
    g1[1] = (int)(tile_w << 16);                         // tensor_dim0[15:0]
    g1[2] = (int)((tile_w >> 16) | (tile_h << 16));      // dim0 hi | tensor_dim1 lo
    g1[3] = (int)((tile_h >> 16) | (tile_w << 16));      // dim1 hi | tile_dim0
    g1[4] = (int)(tile_h & 0xffffu);                     // tile_dim1 (tile_dim2=0)
    g1[5] = (int)stride_elems;                           // tensor_dim0_stride[31:0]
    g1[6] = 0;                                           // stride hi | dim1_stride lo
    g1[7] = 0;
    const v4i z4 = {0, 0, 0, 0};
    const v8i z8 = {0, 0, 0, 0, 0, 0, 0, 0};
    // 6-arg form (this toolchain): (g0, g1, g2, g3, g4, cpol)
    __builtin_amdgcn_tensor_load_to_lds(g0, g1, z4, z4, z8, 0);
}
#endif

// ------------------------------- weight cast --------------------------------
__global__ __launch_bounds__(256)
void cvt_f16_kernel(const float* __restrict__ s, _Float16* __restrict__ d, int n) {
    int i = blockIdx.x * 256 + threadIdx.x;
    if (i < n) d[i] = (_Float16)s[i];
}

// --------------------------- fused layernorm + cast -------------------------
__global__ __launch_bounds__(256)
void ln_cast_kernel(const float* __restrict__ x, const float* __restrict__ w,
                    const float* __restrict__ b, _Float16* __restrict__ y) {
    __shared__ float r1[256], r2[256];
    const int row = blockIdx.x;
    const int tid = threadIdx.x;
    const float* xr = x + (size_t)row * 768;
    float s = 0.f, s2 = 0.f;
    for (int i = tid; i < 768; i += 256) { float v = xr[i]; s += v; s2 += v * v; }
    r1[tid] = s; r2[tid] = s2;
    __syncthreads();
    for (int off = 128; off > 0; off >>= 1) {
        if (tid < off) { r1[tid] += r1[tid + off]; r2[tid] += r2[tid + off]; }
        __syncthreads();
    }
    const float mean = r1[0] * (1.0f / 768.0f);
    const float var  = r2[0] * (1.0f / 768.0f) - mean * mean;
    const float rstd = rsqrtf(var + 1e-5f);
    for (int i = tid; i < 768; i += 256)
        y[(size_t)row * 768 + i] = (_Float16)((xr[i] - mean) * rstd * w[i] + b[i]);
}

// ------------------------------ WMMA GEMM core ------------------------------
// out[M,Nd] = A[M,K] * Bw[Nd,K]^T  (row-major, K contiguous in both).
// Block = 128 threads (4 waves), tile 64x64, K-step 32, TDM-staged LDS tiles.
// MODE 0: QKV scatter (q scaled by HS^-0.5)  MODE 1: proj + bias + residual -> f32
// MODE 2: fc1 + bias + exact GELU -> f16      MODE 3: fc2 + bias + residual -> f32
template <int MODE>
__global__ __launch_bounds__(128)
void gemm_kernel(const _Float16* __restrict__ A, const _Float16* __restrict__ Bw,
                 int K,
                 const float* __restrict__ bias, const float* __restrict__ resid,
                 _Float16* __restrict__ oq, _Float16* __restrict__ ok,
                 _Float16* __restrict__ ov, _Float16* __restrict__ oh,
                 float* __restrict__ of) {
    __shared__ __align__(16) _Float16 sA[64][40];  // 40-half stride (TDM-padded)
    __shared__ __align__(16) _Float16 sB[64][40];

    const int lane  = threadIdx.x & 31;
    const int wv    = threadIdx.x >> 5;
    const int lrow  = lane & 15;
    const int lhalf = lane >> 4;
    const int m0    = blockIdx.x * 64;
    const int n0    = blockIdx.y * 64;

    v8f acc[4];
#pragma unroll
    for (int j = 0; j < 4; ++j)
#pragma unroll
        for (int e = 0; e < 8; ++e) acc[j][e] = 0.f;

    for (int k0 = 0; k0 < K; k0 += 32) {
        __syncthreads();
#if defined(USE_TDM)
        if (wv == 0) {
            // 64x32 f16 tiles; pad 4 DWORDs after each 16 DWORDs (-> 40-half rows)
            tdm_load_2d(lds_offset_of(&sA[0][0]), A  + (size_t)m0 * K + k0,
                        32u, 64u, (unsigned)K, 3u, 3u);
            tdm_load_2d(lds_offset_of(&sB[0][0]), Bw + (size_t)n0 * K + k0,
                        32u, 64u, (unsigned)K, 3u, 3u);
            __builtin_amdgcn_s_wait_tensorcnt(0);
        }
#else
        {
            const int r  = threadIdx.x >> 1;
            const int cs = (threadIdx.x & 1) * 16;
            const _Float16* ga = A  + (size_t)(m0 + r) * K + k0 + cs;
            const _Float16* gb = Bw + (size_t)(n0 + r) * K + k0 + cs;
            *(uint4*)&sA[r][cs]     = *(const uint4*)ga;
            *(uint4*)&sA[r][cs + 8] = *(const uint4*)(ga + 8);
            *(uint4*)&sB[r][cs]     = *(const uint4*)gb;
            *(uint4*)&sB[r][cs + 8] = *(const uint4*)(gb + 8);
            if (k0 + 32 < K) {
                __builtin_prefetch(ga + 32, 0, 1);
                __builtin_prefetch(gb + 32, 0, 1);
            }
        }
#endif
        __syncthreads();

        F16x16 a;                                  // A 16x32 fragment (ISA layout)
        const int ar = wv * 16 + lrow;
        a.q[0] = *(const uint4*)&sA[ar][lhalf * 8];
        a.q[1] = *(const uint4*)&sA[ar][16 + lhalf * 8];
#pragma unroll
        for (int j = 0; j < 4; ++j) {
            F16x16 bf;                             // B 32x16 fragment: lane = col
            const int br = j * 16 + lrow;
            bf.q[0] = *(const uint4*)&sB[br][lhalf * 8];
            bf.q[1] = *(const uint4*)&sB[br][16 + lhalf * 8];
            acc[j] = WMMA_F16(a.v, bf.v, acc[j]);
        }
    }

#pragma unroll
    for (int j = 0; j < 4; ++j) {
#pragma unroll
        for (int i = 0; i < 8; ++i) {
            const int m = m0 + wv * 16 + i + 8 * lhalf;
            const int n = n0 + j * 16 + lrow;
            const float v = acc[j][i];
            if (MODE == 0) {
                const int s   = n / 768;
                const int rem = n - s * 768;
                const int hh  = rem >> 6;
                const int hs  = rem & 63;
                const int bb  = m >> 10;
                const int tk  = m & 1023;
                const size_t idx = (((size_t)bb * 12 + hh) * 1024 + tk) * 64 + hs;
                if      (s == 0) oq[idx] = (_Float16)(v * 0.125f);   // fold SCALE
                else if (s == 1) ok[idx] = (_Float16)v;
                else             ov[idx] = (_Float16)v;
            } else if (MODE == 2) {
                const float x = v + bias[n];
                const float g = 0.5f * x * (1.0f + erff(x * 0.70710678118654752f));
                oh[(size_t)m * 768 + n] = (_Float16)g;
            } else {  // MODE 1 / MODE 3: bias + residual, f32 out
                const size_t idx = (size_t)m * 768 + n;
                of[idx] = v + bias[n] + resid[idx];
            }
        }
    }
}

// ------------------------------- attention ----------------------------------
// grid = (B*H, N/16). Each block: 16-row score strip (16x1024), WMMA QK^T,
// multiplicative k_conn bias, softmax (1/sum folded into AV epilogue),
// WMMA P*V with V staged through LDS by the TDM.
#define ATTN_SMEM (16 * 1024 * 4 + 16 * 1024 * 2 + 128 * 4 + 16 * 4 + 16 * 4)

__global__ __launch_bounds__(128)
void attn_kernel(const _Float16* __restrict__ qh, const _Float16* __restrict__ kh,
                 const _Float16* __restrict__ vh, const float* __restrict__ kc,
                 _Float16* __restrict__ oh) {
    extern __shared__ char smem[];
    float*    S      = (float*)smem;                       // 16x1024 f32 scores
    _Float16* P      = (_Float16*)(smem + 65536);          // 16x1024 f16 exp(scores)
    float*    red    = (float*)(smem + 65536 + 32768);     // 16x8 partials
    float*    rowmax = red + 128;
    float*    rowinv = rowmax + 16;
    _Float16 (*Vt)[72] = (_Float16 (*)[72])smem;           // aliases S (phase 3)

    const int bh    = blockIdx.x;          // 0..95
    const int b     = bh / 12;
    const int h     = bh % 12;
    const int t0    = blockIdx.y * 16;
    const int tid   = threadIdx.x;
    const int lane  = tid & 31;
    const int wv    = tid >> 5;
    const int lrow  = lane & 15;
    const int lhalf = lane >> 4;

    // Q fragments for rows t0..t0+15 (K = hs, two 32-chunks); SCALE pre-folded.
    const _Float16* qrow = qh + ((size_t)bh * 1024 + t0 + lrow) * 64;
    F16x16 a0, a1;
    a0.q[0] = *(const uint4*)(qrow + lhalf * 8);
    a0.q[1] = *(const uint4*)(qrow + 16 + lhalf * 8);
    a1.q[0] = *(const uint4*)(qrow + 32 + lhalf * 8);
    a1.q[1] = *(const uint4*)(qrow + 48 + lhalf * 8);

    // ---- Phase 1: scores = (q*SCALE)·k^T, then * k_conn ----
    for (int cj = 0; cj < 16; ++cj) {
        const int col0 = wv * 256 + cj * 16;
        const _Float16* krow = kh + ((size_t)bh * 1024 + col0 + lrow) * 64;
        v8f acc;
#pragma unroll
        for (int e = 0; e < 8; ++e) acc[e] = 0.f;
        F16x16 b0, b1;
        b0.q[0] = *(const uint4*)(krow + lhalf * 8);
        b0.q[1] = *(const uint4*)(krow + 16 + lhalf * 8);
        acc = WMMA_F16(a0.v, b0.v, acc);
        b1.q[0] = *(const uint4*)(krow + 32 + lhalf * 8);
        b1.q[1] = *(const uint4*)(krow + 48 + lhalf * 8);
        acc = WMMA_F16(a1.v, b1.v, acc);
#pragma unroll
        for (int i = 0; i < 8; ++i) {
            const int mr = i + 8 * lhalf;
            const int n  = col0 + lrow;
            S[mr * 1024 + n] = acc[i] * kc[((size_t)b * 1024 + t0 + mr) * 1024 + n];
        }
    }
    __syncthreads();

    // ---- Phase 2: row softmax (numerator only; 1/sum deferred) ----
    const int r  = tid >> 3;   // 0..15
    const int c8 = tid & 7;    // 0..7
    {
        float mx = -3.0e38f;
        for (int j = c8 * 128; j < c8 * 128 + 128; ++j) mx = fmaxf(mx, S[r * 1024 + j]);
        red[r * 8 + c8] = mx;
    }
    __syncthreads();
    if (c8 == 0) {
        float mx = red[r * 8];
        for (int j = 1; j < 8; ++j) mx = fmaxf(mx, red[r * 8 + j]);
        rowmax[r] = mx;
    }
    __syncthreads();
    {
        const float rm = rowmax[r];
        float sum = 0.f;
        for (int j = c8 * 128; j < c8 * 128 + 128; ++j) {
            const float e = __expf(S[r * 1024 + j] - rm);
            P[r * 1024 + j] = (_Float16)e;
            sum += e;
        }
        red[r * 8 + c8] = sum;
    }
    __syncthreads();
    if (c8 == 0) {
        float s = 0.f;
        for (int j = 0; j < 8; ++j) s += red[r * 8 + j];
        rowinv[r] = 1.0f / s;
    }

    // ---- Phase 3: out(16x64) = P(16x1024) x V(1024x64) ----
    v8f acc;
#pragma unroll
    for (int e = 0; e < 8; ++e) acc[e] = 0.f;
    for (int k0 = 0; k0 < 1024; k0 += 32) {
        __syncthreads();               // also fences rowinv + retires prior reads
#if defined(USE_TDM)
        if (wv == 0) {
            // 32x64 f16 V tile; pad 4 DWORDs after each 32 DWORDs (-> 72-half rows)
            tdm_load_2d(lds_offset_of(smem),
                        vh + ((size_t)bh * 1024 + k0) * 64,
                        64u, 32u, 64u, 4u, 3u);
            __builtin_amdgcn_s_wait_tensorcnt(0);
        }
#else
        {
            const int vr = tid >> 2;           // 0..31
            const int vs = (tid & 3) * 16;     // 0,16,32,48
            const _Float16* gv = vh + ((size_t)bh * 1024 + k0 + vr) * 64 + vs;
            *(uint4*)&Vt[vr][vs]     = *(const uint4*)gv;
            *(uint4*)&Vt[vr][vs + 8] = *(const uint4*)(gv + 8);
        }
#endif
        __syncthreads();
        F16x16 pa, vb;
        pa.q[0] = *(const uint4*)&P[lrow * 1024 + k0 + lhalf * 8];
        pa.q[1] = *(const uint4*)&P[lrow * 1024 + k0 + 16 + lhalf * 8];
        const int hs = wv * 16 + lrow;
        const int kb = lhalf * 8;
#pragma unroll
        for (int kk = 0; kk < 16; ++kk) vb.h[kk] = Vt[kb + (kk & 7) + (kk >> 3) * 16][hs];  // ds gathers
        acc = WMMA_F16(pa.v, vb.v, acc);
    }
#pragma unroll
    for (int i = 0; i < 8; ++i) {
        const int mr = i + 8 * lhalf;
        const int hs = wv * 16 + lrow;
        oh[((size_t)b * 1024 + t0 + mr) * 768 + h * 64 + hs] =
            (_Float16)(acc[i] * rowinv[mr]);
    }
}

// ------------------------------- launcher -----------------------------------
extern "C" void kernel_launch(void* const* d_in, const int* in_sizes, int n_in,
                              void* d_out, int out_size, void* d_ws, size_t ws_size,
                              hipStream_t stream) {
    (void)in_sizes; (void)n_in; (void)out_size; (void)ws_size;
    const float* jf     = (const float*)d_in[0];
    const float* k_conn = (const float*)d_in[1];
    const float* qkv_w  = (const float*)d_in[2];
    const float* proj_w = (const float*)d_in[3];
    const float* proj_b = (const float*)d_in[4];
    const float* fc1_w  = (const float*)d_in[5];
    const float* fc1_b  = (const float*)d_in[6];
    const float* fc2_w  = (const float*)d_in[7];
    const float* fc2_b  = (const float*)d_in[8];
    const float* ln1_w  = (const float*)d_in[9];
    const float* ln1_b  = (const float*)d_in[10];
    const float* ln2_w  = (const float*)d_in[11];
    const float* ln2_b  = (const float*)d_in[12];
    float* out = (float*)d_out;

    // ---- workspace layout (~120 MB) ----
    char* ws = (char*)d_ws;
    const size_t SZ_WQ = (size_t)2304 * 768 * 2;
    const size_t SZ_WC = (size_t)768 * 768 * 2;
    const size_t SZ_XH = (size_t)8192 * 768 * 2;
    const size_t SZ_YF = (size_t)8192 * 768 * 4;
    size_t off = 0;
    _Float16* Wq  = (_Float16*)(ws + off); off += SZ_WQ;
    _Float16* Wp  = (_Float16*)(ws + off); off += SZ_WC;
    _Float16* W1  = (_Float16*)(ws + off); off += SZ_WC;
    _Float16* W2  = (_Float16*)(ws + off); off += SZ_WC;
    _Float16* xh  = (_Float16*)(ws + off); off += SZ_XH;
    _Float16* qhp = (_Float16*)(ws + off); off += SZ_XH;
    _Float16* khp = (_Float16*)(ws + off); off += SZ_XH;
    _Float16* vhp = (_Float16*)(ws + off); off += SZ_XH;
    _Float16* ah  = (_Float16*)(ws + off); off += SZ_XH;
    float*    yf  = (float*)(ws + off);    off += SZ_YF;
    _Float16* zh  = (_Float16*)(ws + off); off += SZ_XH;
    _Float16* hh  = (_Float16*)(ws + off); off += SZ_XH;

    // 1) weights -> f16 (once per call; cheap vs GEMMs)
    cvt_f16_kernel<<<(2304 * 768 + 255) / 256, 256, 0, stream>>>(qkv_w, Wq, 2304 * 768);
    cvt_f16_kernel<<<(768 * 768 + 255) / 256, 256, 0, stream>>>(proj_w, Wp, 768 * 768);
    cvt_f16_kernel<<<(768 * 768 + 255) / 256, 256, 0, stream>>>(fc1_w, W1, 768 * 768);
    cvt_f16_kernel<<<(768 * 768 + 255) / 256, 256, 0, stream>>>(fc2_w, W2, 768 * 768);

    // 2) LN1 + cast
    ln_cast_kernel<<<8192, 256, 0, stream>>>(jf, ln1_w, ln1_b, xh);

    // 3) QKV GEMM (8192 x 2304 x 768) -> per-head q/k/v f16
    gemm_kernel<0><<<dim3(128, 36), 128, 0, stream>>>(
        xh, Wq, 768, nullptr, nullptr, qhp, khp, vhp, nullptr, nullptr);

    // 4) attention (96 heads x 64 row-tiles)
    attn_kernel<<<dim3(96, 64), 128, ATTN_SMEM, stream>>>(qhp, khp, vhp, k_conn, ah);

    // 5) proj GEMM + bias + residual -> y (f32)
    gemm_kernel<1><<<dim3(128, 12), 128, 0, stream>>>(
        ah, Wp, 768, proj_b, jf, nullptr, nullptr, nullptr, nullptr, yf);

    // 6) LN2 + cast
    ln_cast_kernel<<<8192, 256, 0, stream>>>(yf, ln2_w, ln2_b, zh);

    // 7) FC1 + bias + exact GELU -> f16
    gemm_kernel<2><<<dim3(128, 12), 128, 0, stream>>>(
        zh, W1, 768, fc1_b, nullptr, nullptr, nullptr, nullptr, hh, nullptr);

    // 8) FC2 + bias + residual(y) -> out (f32)
    gemm_kernel<3><<<dim3(128, 12), 128, 0, stream>>>(
        hh, W2, 768, fc2_b, yf, nullptr, nullptr, nullptr, nullptr, out);
}
